// GCGRU_60163901883015
// MI455X (gfx1250) — matrix operand order repaired and checked by reading, not compiled, not run on hardware
//
#include <hip/hip_runtime.h>
#include <math.h>

#define DEV __device__ __forceinline__

typedef __attribute__((ext_vector_type(16))) __bf16 v16bf;
typedef __attribute__((ext_vector_type(8)))  float  v8f;

union Frag { v16bf v; uint4 q[2]; };

DEV unsigned short f2bf(float f) {
  unsigned int u = __float_as_uint(f);
  u += 0x7FFFu + ((u >> 16) & 1u);          // round-to-nearest-even
  return (unsigned short)(u >> 16);
}
DEV float sigf(float x) { return 1.0f / (1.0f + expf(-x)); }

// ---- problem constants ----
static constexpr int Bn = 16, Pn = 15, Nn = 1000, Fn = 8, Cn = 64, NP = 1024;
static constexpr int Rg = Pn * Bn * Nn;   // 240000 rows of (p,b,n)
static constexpr int R2 = Bn * Nn;        // 16000 rows of (b,n)

// ---- gfx1250 async-to-LDS path (probe via __has_builtin; sync fallback) ----
#if defined(__has_builtin)
#if __has_builtin(__builtin_amdgcn_global_load_async_to_lds_b128) && \
    __has_builtin(__builtin_amdgcn_s_wait_asynccnt)
#define USE_ASYNC_LDS 1
#endif
#endif
#ifndef USE_ASYNC_LDS
#define USE_ASYNC_LDS 0
#endif

#if USE_ASYNC_LDS
// Builtin signature (from hipcc diagnostics): param types are
// 'int __vector_size__(16) *' in AS1 (global) / AS3 (LDS).
typedef int v4i __attribute__((vector_size(16)));
typedef __attribute__((address_space(1))) v4i* glb_v4i;
typedef __attribute__((address_space(3))) v4i* lds_v4i;
// AS1 value == flat value. Flat pointers to LDS carry the DS offset in the
// low 32 bits (aperture in high bits), so trunc-to-32 recovers AS3.
DEV glb_v4i to_glb(const unsigned short* p) {
  return (glb_v4i)(unsigned long long)p;
}
DEV lds_v4i to_lds(unsigned short* p) {
  return (lds_v4i)(unsigned int)(unsigned long long)p;
}
#endif

// =====================================================================
// Universal bf16 WMMA GEMM: C[M,Nc] = act(A[M,K] * B^T[Nc,K] + bias)
// A row-major K-contiguous (lda), B stored transposed [col][K] (ldb).
// Block: 256 thr = 8 waves (4M x 2N); block tile 128x64; wave tile 32x32.
// K consumed in steps of 32 (one v_wmma_f32_16x16x32_bf16 per step/tile).
// Double-buffered LDS; tiles filled by GLOBAL_LOAD_ASYNC_TO_LDS_B128
// (ASYNCcnt) when available, overlapping the WMMA consumption of the
// other buffer; one workgroup barrier per K-step.
// All dims pre-padded: M%128==0, Nc%64==0, K%32==0, no bounds checks.
// =====================================================================
__global__ __launch_bounds__(256)
void k_gemm(const unsigned short* __restrict__ A, long lda, long strideA,
            const unsigned short* __restrict__ Bt, long ldb, long strideB,
            void* __restrict__ Cp, long ldc, long strideC, int outBf16,
            int nK, const float* __restrict__ bias, int act)
{
  __shared__ unsigned short As[2][128 * 40];   // 128 rows x 32 K (+8 pad)
  __shared__ unsigned short Bs[2][64 * 40];    // 64 cols x 32 K (+8 pad)

  const int tid  = threadIdx.x;
  const int lane = tid & 31;
  const int w    = tid >> 5;
  const int wm   = w & 3;          // 0..3  -> M offset wm*32
  const int wn   = w >> 2;         // 0..1  -> N offset wn*32
  const int lm   = lane & 15;
  const int l16  = lane >> 4;

  const long blockM = (long)blockIdx.x * 128;
  const long blockN = (long)blockIdx.y * 64;
  const unsigned short* Ab = A  + (size_t)blockIdx.z * (size_t)strideA;
  const unsigned short* Bb = Bt + (size_t)blockIdx.z * (size_t)strideB;

  v8f acc[2][2];
#pragma unroll
  for (int i = 0; i < 2; ++i)
#pragma unroll
    for (int j = 0; j < 2; ++j) acc[i][j] = (v8f)0.0f;

#if USE_ASYNC_LDS
  // ---- async double-buffered pipeline ----
  auto issue = [&](int buf, int kk) {
#pragma unroll
    for (int i = 0; i < 2; ++i) {            // A: 512 16B chunks / 256 thr
      int ch = tid + i * 256;
      int r = ch >> 2, c = ch & 3;
      __builtin_amdgcn_global_load_async_to_lds_b128(
          to_glb(Ab + (size_t)(blockM + r) * lda + (size_t)kk * 32 + c * 8),
          to_lds(&As[buf][r * 40 + c * 8]), 0, 0);
    }
    {                                        // B: 256 16B chunks / 256 thr
      int r = tid >> 2, c = tid & 3;
      __builtin_amdgcn_global_load_async_to_lds_b128(
          to_glb(Bb + (size_t)(blockN + r) * ldb + (size_t)kk * 32 + c * 8),
          to_lds(&Bs[buf][r * 40 + c * 8]), 0, 0);
    }
  };
  issue(0, 0);
  for (int kk = 0; kk < nK; ++kk) {
    const int buf = kk & 1;
    __builtin_amdgcn_s_wait_asynccnt(0);     // own tile-kk loads landed
    __syncthreads();                         // everyone's tile-kk landed
    if (kk + 1 < nK) issue(buf ^ 1, kk + 1); // fill other buffer during WMMA

    Frag af[2], bfr[2];
#pragma unroll
    for (int mt = 0; mt < 2; ++mt) {
      const unsigned short* ar = &As[buf][(wm * 32 + mt * 16 + lm) * 40];
      af[mt].q[0] = *(const uint4*)(ar + 16 * l16);
      af[mt].q[1] = *(const uint4*)(ar + 16 * l16 + 8);
    }
#pragma unroll
    for (int nt = 0; nt < 2; ++nt) {
      const unsigned short* br = &Bs[buf][(wn * 32 + nt * 16 + lm) * 40 + 16 * l16];
      bfr[nt].q[0] = *(const uint4*)(br);
      bfr[nt].q[1] = *(const uint4*)(br + 8);
    }
#pragma unroll
    for (int mt = 0; mt < 2; ++mt)
#pragma unroll
      for (int nt = 0; nt < 2; ++nt)
        acc[mt][nt] = __builtin_amdgcn_wmma_f32_16x16x32_bf16(
            false, af[mt].v, false, bfr[nt].v, (short)0, acc[mt][nt],
            false, false);
  }
#else
  // ---- sync fallback: VGPR-staged global->LDS ----
  for (int kk = 0; kk < nK; ++kk) {
#pragma unroll
    for (int i = 0; i < 2; ++i) {
      int ch = tid + i * 256;
      int r = ch >> 2, c = ch & 3;
      const uint4* s = (const uint4*)(Ab + (size_t)(blockM + r) * lda
                                         + (size_t)kk * 32 + c * 8);
      *((uint4*)&As[0][r * 40 + c * 8]) = *s;
    }
    {
      int r = tid >> 2, c = tid & 3;
      const uint4* s = (const uint4*)(Bb + (size_t)(blockN + r) * ldb
                                         + (size_t)kk * 32 + c * 8);
      *((uint4*)&Bs[0][r * 40 + c * 8]) = *s;
    }
    __syncthreads();

    Frag af[2], bfr[2];
#pragma unroll
    for (int mt = 0; mt < 2; ++mt) {
      const unsigned short* ar = &As[0][(wm * 32 + mt * 16 + lm) * 40];
      af[mt].q[0] = *(const uint4*)(ar + 16 * l16);
      af[mt].q[1] = *(const uint4*)(ar + 16 * l16 + 8);
    }
#pragma unroll
    for (int nt = 0; nt < 2; ++nt) {
      const unsigned short* br = &Bs[0][(wn * 32 + nt * 16 + lm) * 40 + 16 * l16];
      bfr[nt].q[0] = *(const uint4*)(br);
      bfr[nt].q[1] = *(const uint4*)(br + 8);
    }
#pragma unroll
    for (int mt = 0; mt < 2; ++mt)
#pragma unroll
      for (int nt = 0; nt < 2; ++nt)
        acc[mt][nt] = __builtin_amdgcn_wmma_f32_16x16x32_bf16(
            false, af[mt].v, false, bfr[nt].v, (short)0, acc[mt][nt],
            false, false);
    __syncthreads();
  }
#endif

  // ---- epilogue: bias + activation, f32 or bf16 store ----
  unsigned short* C16 = (unsigned short*)Cp + (size_t)blockIdx.z * (size_t)strideC;
  float*          C32 = (float*)Cp          + (size_t)blockIdx.z * (size_t)strideC;
#pragma unroll
  for (int mt = 0; mt < 2; ++mt)
#pragma unroll
    for (int nt = 0; nt < 2; ++nt) {
      long colg = blockN + wn * 32 + nt * 16 + lm;
      float bv = bias ? bias[colg] : 0.0f;
#pragma unroll
      for (int r = 0; r < 8; ++r) {
        long rowg = blockM + wm * 32 + mt * 16 + r + 8 * l16;
        float v = acc[mt][nt][r] + bv;
        if (act == 1) v = tanhf(v);
        else if (act == 2) v = sigf(v);
        if (outBf16) C16[(size_t)rowg * ldc + colg] = f2bf(v);
        else         C32[(size_t)rowg * ldc + colg] = v;
      }
    }
}

// =====================================================================
// Pack / elementwise kernels
// =====================================================================
__global__ void k_pack_mats(const float* __restrict__ m, unsigned short* __restrict__ dst) {
  int idx = blockIdx.x * 256 + threadIdx.x;        // 1024*1024
  int n = idx >> 10, c = idx & 1023;
  float v = (n < Nn && c < Nn) ? m[n * Nn + c] : 0.0f;
  dst[idx] = f2bf(v);
}

__global__ void k_pack_adj(const float* __restrict__ adj, unsigned short* __restrict__ dst) {
  long idx = (long)blockIdx.x * 256 + threadIdx.x; // 15*1024*1024
  int p = (int)(idx >> 20);
  int rem = (int)(idx & 1048575);
  int n = rem >> 10, c = rem & 1023;
  float v = 0.0f;
  if (n < Nn && c < Nn)   // adj[p][p][n][c]; read-once slice of a 960MB tensor
    v = __builtin_nontemporal_load(&adj[(size_t)p * 16000000 + (size_t)n * Nn + c]);
  dst[idx] = f2bf(v);
}

__global__ void k_pack_xBT(const float* __restrict__ x, unsigned short* __restrict__ dst) {
  int idx = blockIdx.x * 256 + threadIdx.x;        // 1920*1024
  int j = idx >> 10, m = idx & 1023;
  int p = j >> 7, rem = j & 127, b = rem >> 3, f = rem & 7; // j=(p*B+b)*F+f
  float v = (m < Nn) ? x[(((size_t)b * Pn + p) * Nn + m) * Fn + f] : 0.0f;
  dst[idx] = f2bf(v);
}

__global__ void k_pack_T0_rows(const unsigned short* __restrict__ T0,
                               unsigned short* __restrict__ dst) {
  long idx = (long)blockIdx.x * 256 + threadIdx.x; // Rg*32
  int r = (int)(idx >> 5), k = (int)(idx & 31);
  int p = r / (Bn * Nn), rem = r % (Bn * Nn), b = rem / Nn, n = rem % Nn;
  unsigned short v = 0;
  if (k < Fn) v = T0[(size_t)n * 1920 + (p * Bn + b) * Fn + k];
  dst[idx] = v;
}

__global__ void k_pack_G_BT(const unsigned short* __restrict__ G,
                            unsigned short* __restrict__ dst) {
  long idx = (long)blockIdx.x * 256 + threadIdx.x; // 15360*1024
  int j = (int)(idx >> 10), m = (int)(idx & 1023);
  unsigned short v = 0;
  if (m < Nn) {
    int pb = j >> 6, c = j & 63;                   // j=(p*B+b)*64+c
    v = G[((size_t)pb * Nn + m) * 64 + c];
  }
  dst[idx] = v;
}

__global__ void k_pack_T2_rows(const unsigned short* __restrict__ T2,
                               unsigned short* __restrict__ dst) {
  long idx = (long)blockIdx.x * 256 + threadIdx.x; // Rg*64
  int r = (int)(idx >> 6), c = (int)(idx & 63);
  int p = r / (Bn * Nn), rem = r % (Bn * Nn), b = rem / Nn, n = rem % Nn;
  dst[idx] = T2[(size_t)n * 15360 + (p * Bn + b) * 64 + c];
}

__global__ void k_pack_U0_rows(const unsigned short* __restrict__ U0,
                               unsigned short* __restrict__ dst) {
  long idx = (long)blockIdx.x * 256 + threadIdx.x; // Rg*32
  int r = (int)(idx >> 5), k = (int)(idx & 31);
  int p = r / (Bn * Nn), rem = r % (Bn * Nn), b = rem / Nn, n = rem % Nn;
  unsigned short v = 0;
  if (k < Fn) v = U0[((size_t)p * NP + n) * 128 + b * Fn + k];
  dst[idx] = v;
}

__global__ void k_pack_U2_rows(const unsigned short* __restrict__ U2,
                               unsigned short* __restrict__ dst) {
  long idx = (long)blockIdx.x * 256 + threadIdx.x; // Rg*64
  int r = (int)(idx >> 6), c = (int)(idx & 63);
  int p = r / (Bn * Nn), rem = r % (Bn * Nn), b = rem / Nn, n = rem % Nn;
  dst[idx] = U2[((size_t)p * NP + n) * 1024 + b * 64 + c];
}

__global__ void k_pack_wT(const float* __restrict__ W, int Kin, int Cout,
                          unsigned short* __restrict__ dst, int Kpad) {
  int idx = blockIdx.x * 256 + threadIdx.x;
  if (idx >= Cout * Kpad) return;
  int c = idx / Kpad, k = idx % Kpad;
  dst[idx] = (k < Kin) ? f2bf(W[k * Cout + c]) : (unsigned short)0;
}

__global__ void k_concat_bias(float* __restrict__ dst, const float* __restrict__ a,
                              const float* __restrict__ b) {
  int i = threadIdx.x;          // 64 threads
  dst[i] = a[i];
  dst[64 + i] = b[i];
}

__global__ void k_pack_Z0(const float* __restrict__ x,
                          const unsigned short* __restrict__ Gf,
                          const unsigned short* __restrict__ G1f,
                          const float* __restrict__ H0,
                          unsigned short* __restrict__ Z, int t) {
  long idx = (long)blockIdx.x * 256 + threadIdx.x; // R2*224
  int r2 = (int)(idx / 224), k = (int)(idx % 224);
  int b = r2 / Nn, n = r2 % Nn;
  unsigned short v = 0;
  if (k < 8)        v = f2bf(x[(((size_t)b * Pn + t) * Nn + n) * Fn + k]);
  else if (k < 72)  v = Gf [(((size_t)t * Bn + b) * Nn + n) * 64 + (k - 8)];
  else if (k < 136) v = G1f[(((size_t)t * Bn + b) * Nn + n) * 64 + (k - 72)];
  else if (k < 200) v = f2bf(H0[(size_t)r2 * 64 + (k - 136)]);
  Z[idx] = v;
}

__global__ void k_patch_Zc0(unsigned short* __restrict__ Z,
                            const float* __restrict__ UR,
                            const float* __restrict__ H) {
  long idx = (long)blockIdx.x * 256 + threadIdx.x; // R2*64
  int r2 = (int)(idx >> 6), c = (int)(idx & 63);
  Z[(size_t)r2 * 224 + 136 + c] = f2bf(UR[(size_t)r2 * 128 + 64 + c] * H[idx]);
}

__global__ void k_pack_Z1(const unsigned short* __restrict__ Gf,
                          const unsigned short* __restrict__ G1f,
                          const float* __restrict__ H0,
                          const float* __restrict__ H1,
                          unsigned short* __restrict__ Z, int t) {
  long idx = (long)blockIdx.x * 256 + threadIdx.x; // R2*256
  int r2 = (int)(idx >> 8), k = (int)(idx & 255);
  int b = r2 / Nn, n = r2 % Nn;
  unsigned short v;
  if (k < 64)       v = f2bf(H0[(size_t)r2 * 64 + k]);
  else if (k < 128) v = Gf [(((size_t)t * Bn + b) * Nn + n) * 64 + (k - 64)];
  else if (k < 192) v = G1f[(((size_t)t * Bn + b) * Nn + n) * 64 + (k - 128)];
  else              v = f2bf(H1[(size_t)r2 * 64 + (k - 192)]);
  Z[idx] = v;
}

__global__ void k_patch_Zc1(unsigned short* __restrict__ Z,
                            const float* __restrict__ UR,
                            const float* __restrict__ H) {
  long idx = (long)blockIdx.x * 256 + threadIdx.x; // R2*64
  int r2 = (int)(idx >> 6), c = (int)(idx & 63);
  Z[(size_t)r2 * 256 + 192 + c] = f2bf(UR[(size_t)r2 * 128 + 64 + c] * H[idx]);
}

__global__ void k_update_h(float* __restrict__ H, const float* __restrict__ UR,
                           const float* __restrict__ Cc) {
  long idx = (long)blockIdx.x * 256 + threadIdx.x; // R2*64
  int r2 = (int)(idx >> 6), c = (int)(idx & 63);
  float u = UR[(size_t)r2 * 128 + c];
  H[idx] = u * H[idx] + (1.0f - u) * Cc[idx];
}

__global__ void k_head(const float* __restrict__ H1,
                       const float* __restrict__ wo, const float* __restrict__ bo,
                       const float* __restrict__ wf0, const float* __restrict__ bf0,
                       const float* __restrict__ wf1, const float* __restrict__ bf1,
                       float* __restrict__ out) {
  int r2 = blockIdx.x * 256 + threadIdx.x;
  if (r2 >= R2) return;
  float h[64];
#pragma unroll
  for (int k = 0; k < 64; ++k) h[k] = H1[(size_t)r2 * 64 + k];
  float xg[32];
  for (int g = 0; g < 32; ++g) {
    float s = bo[g];
#pragma unroll
    for (int k = 0; k < 64; ++k) s += h[k] * wo[k * 32 + g];
    xg[g] = sigf(s);
  }
  float accf = bf1[0];
  for (int j = 0; j < 16; ++j) {
    float s = bf0[j];
#pragma unroll
    for (int g = 0; g < 32; ++g) s += xg[g] * wf0[g * 16 + j];
    accf += wf1[j] * sigf(s);
  }
  out[r2] = sigf(accf);
}

// =====================================================================
// Host orchestration
// =====================================================================
extern "C" void kernel_launch(void* const* d_in, const int* in_sizes, int n_in,
                              void* d_out, int out_size, void* d_ws, size_t ws_size,
                              hipStream_t stream) {
  const float* x    = (const float*)d_in[0];
  const float* adj  = (const float*)d_in[1];
  const float* mats = (const float*)d_in[2];
  const float* wg0 = (const float*)d_in[3];  const float* bg0 = (const float*)d_in[4];
  const float* wg1 = (const float*)d_in[5];  const float* bg1 = (const float*)d_in[6];
  const float* wh0 = (const float*)d_in[7];  const float* bh0 = (const float*)d_in[8];
  const float* wh1 = (const float*)d_in[9];  const float* bh1 = (const float*)d_in[10];
  const float* wu0 = (const float*)d_in[11]; const float* bu0 = (const float*)d_in[12];
  const float* wr0 = (const float*)d_in[13]; const float* br0 = (const float*)d_in[14];
  const float* wc0 = (const float*)d_in[15]; const float* bc0 = (const float*)d_in[16];
  const float* wu1 = (const float*)d_in[17]; const float* bu1 = (const float*)d_in[18];
  const float* wr1 = (const float*)d_in[19]; const float* br1 = (const float*)d_in[20];
  const float* wc1 = (const float*)d_in[21]; const float* bc1 = (const float*)d_in[22];
  const float* wo  = (const float*)d_in[23]; const float* bo  = (const float*)d_in[24];
  const float* wf0 = (const float*)d_in[25]; const float* bf0 = (const float*)d_in[26];
  const float* wf1 = (const float*)d_in[27]; const float* bf1 = (const float*)d_in[28];
  (void)in_sizes; (void)n_in; (void)out_size; (void)ws_size;

  char* wsb = (char*)d_ws;
  size_t off = 0;
  auto alloc = [&](size_t bytes) -> char* {
    char* p = wsb + off;
    off += (bytes + 255) & ~(size_t)255;
    return p;
  };

  unsigned short* Abf   = (unsigned short*)alloc(2ull * NP * NP);          // mats bf16
  unsigned short* Aadj  = (unsigned short*)alloc(2ull * Pn * NP * NP);     // A_p bf16
  unsigned short* BTx   = (unsigned short*)alloc(2ull * 1920 * NP);        // x^T (cols=(p,b,f))
  unsigned short* R1    = (unsigned short*)alloc(2ull * Pn * NP * NP);     // T0/T2/U0/U2 (reused)
  unsigned short* Gtmp  = (unsigned short*)alloc(2ull * Rg * 64);          // G1 / G1b
  unsigned short* BTg   = (unsigned short*)alloc(2ull * 15360 * NP);       // BT_g1 / BT_g1b
  unsigned short* Arows = (unsigned short*)alloc(2ull * Rg * 64);          // dense-GEMM A operand
  unsigned short* Gfin  = (unsigned short*)alloc(2ull * Rg * 64);          // g  [persist]
  unsigned short* G1fin = (unsigned short*)alloc(2ull * Rg * 64);          // g1 [persist]
  unsigned short* Z0    = (unsigned short*)alloc(2ull * R2 * 224);
  unsigned short* Z1    = (unsigned short*)alloc(2ull * R2 * 256);
  float* UR0 = (float*)alloc(4ull * R2 * 128);
  float* UR1 = (float*)alloc(4ull * R2 * 128);
  float* Cc0 = (float*)alloc(4ull * R2 * 64);
  float* Cc1 = (float*)alloc(4ull * R2 * 64);
  float* H0s = (float*)alloc(4ull * R2 * 64);
  float* H1s = (float*)alloc(4ull * R2 * 64);
  unsigned short* WurT0 = (unsigned short*)alloc(2ull * 128 * 224);
  unsigned short* WcT0  = (unsigned short*)alloc(2ull * 64 * 224);
  unsigned short* WurT1 = (unsigned short*)alloc(2ull * 128 * 256);
  unsigned short* WcT1  = (unsigned short*)alloc(2ull * 64 * 256);
  unsigned short* Wg0T  = (unsigned short*)alloc(2ull * 64 * 32);
  unsigned short* Wg1T  = (unsigned short*)alloc(2ull * 64 * 64);
  unsigned short* Wh0T  = (unsigned short*)alloc(2ull * 64 * 32);
  unsigned short* Wh1T  = (unsigned short*)alloc(2ull * 64 * 64);
  float* biasUR0 = (float*)alloc(4ull * 128);
  float* biasUR1 = (float*)alloc(4ull * 128);

  // ---- weight packs (cheap, redone every call; deterministic) ----
  k_pack_wT<<<(64 * 32 + 255) / 256, 256, 0, stream>>>(wg0, 8, 64, Wg0T, 32);
  k_pack_wT<<<(64 * 64 + 255) / 256, 256, 0, stream>>>(wg1, 64, 64, Wg1T, 64);
  k_pack_wT<<<(64 * 32 + 255) / 256, 256, 0, stream>>>(wh0, 8, 64, Wh0T, 32);
  k_pack_wT<<<(64 * 64 + 255) / 256, 256, 0, stream>>>(wh1, 64, 64, Wh1T, 64);
  k_pack_wT<<<(64 * 224 + 255) / 256, 256, 0, stream>>>(wu0, 200, 64, WurT0, 224);
  k_pack_wT<<<(64 * 224 + 255) / 256, 256, 0, stream>>>(wr0, 200, 64, WurT0 + 64 * 224, 224);
  k_pack_wT<<<(64 * 224 + 255) / 256, 256, 0, stream>>>(wc0, 200, 64, WcT0, 224);
  k_pack_wT<<<(64 * 256 + 255) / 256, 256, 0, stream>>>(wu1, 256, 64, WurT1, 256);
  k_pack_wT<<<(64 * 256 + 255) / 256, 256, 0, stream>>>(wr1, 256, 64, WurT1 + 64 * 256, 256);
  k_pack_wT<<<(64 * 256 + 255) / 256, 256, 0, stream>>>(wc1, 256, 64, WcT1, 256);
  k_concat_bias<<<1, 64, 0, stream>>>(biasUR0, bu0, br0);
  k_concat_bias<<<1, 64, 0, stream>>>(biasUR1, bu1, br1);

  // ---- Stage A: fixed-matrix GCN ----
  k_pack_mats<<<4096, 256, 0, stream>>>(mats, Abf);
  k_pack_xBT<<<7680, 256, 0, stream>>>(x, BTx);
  // T0 = mats @ X  -> [1024][1920] bf16
  k_gemm<<<dim3(8, 30, 1), 256, 0, stream>>>(Abf, NP, 0, BTx, NP, 0,
                                             R1, 1920, 0, 1, 32, nullptr, 0);
  k_pack_T0_rows<<<30000, 256, 0, stream>>>(R1, Arows);
  // G1 = tanh(T0 @ wg0 + bg0)
  k_gemm<<<dim3(1875, 1, 1), 256, 0, stream>>>(Arows, 32, 0, Wg0T, 32, 0,
                                               Gtmp, 64, 0, 1, 1, bg0, 1);
  k_pack_G_BT<<<61440, 256, 0, stream>>>(Gtmp, BTg);
  // T2 = mats @ G1 -> [1024][15360] bf16
  k_gemm<<<dim3(8, 240, 1), 256, 0, stream>>>(Abf, NP, 0, BTg, NP, 0,
                                              R1, 15360, 0, 1, 32, nullptr, 0);
  k_pack_T2_rows<<<60000, 256, 0, stream>>>(R1, Arows);
  // Gfin = tanh(T2 @ wg1 + bg1)
  k_gemm<<<dim3(1875, 1, 1), 256, 0, stream>>>(Arows, 64, 0, Wg1T, 64, 0,
                                               Gfin, 64, 0, 1, 2, bg1, 1);

  // ---- Stage B: dynamic-adjacency GCN (batched over p) ----
  k_pack_adj<<<61440, 256, 0, stream>>>(adj, Aadj);
  // U0[p] = A_p @ X_p -> [15][1024][128] bf16
  k_gemm<<<dim3(8, 2, 15), 256, 0, stream>>>(Aadj, NP, (long)NP * NP,
                                             BTx, NP, 128l * NP,
                                             R1, 128, 1024l * 128, 1, 32, nullptr, 0);
  k_pack_U0_rows<<<30000, 256, 0, stream>>>(R1, Arows);
  k_gemm<<<dim3(1875, 1, 1), 256, 0, stream>>>(Arows, 32, 0, Wh0T, 32, 0,
                                               Gtmp, 64, 0, 1, 1, bh0, 1);
  k_pack_G_BT<<<61440, 256, 0, stream>>>(Gtmp, BTg);
  // U2[p] = A_p @ G1b_p -> [15][1024][1024] bf16
  k_gemm<<<dim3(8, 16, 15), 256, 0, stream>>>(Aadj, NP, (long)NP * NP,
                                              BTg, NP, (long)NP * NP,
                                              R1, 1024, (long)NP * NP, 1, 32, nullptr, 0);
  k_pack_U2_rows<<<60000, 256, 0, stream>>>(R1, Arows);
  k_gemm<<<dim3(1875, 1, 1), 256, 0, stream>>>(Arows, 64, 0, Wh1T, 64, 0,
                                               G1fin, 64, 0, 1, 2, bh1, 1);

  // ---- Stage C: 2-layer GRU over 15 timesteps ----
  (void)hipMemsetAsync(H0s, 0, 4ull * R2 * 64, stream);
  (void)hipMemsetAsync(H1s, 0, 4ull * R2 * 64, stream);
  for (int t = 0; t < Pn; ++t) {
    k_pack_Z0<<<14000, 256, 0, stream>>>(x, Gfin, G1fin, H0s, Z0, t);
    k_gemm<<<dim3(125, 2, 1), 256, 0, stream>>>(Z0, 224, 0, WurT0, 224, 0,
                                                UR0, 128, 0, 0, 7, biasUR0, 2);
    k_patch_Zc0<<<4000, 256, 0, stream>>>(Z0, UR0, H0s);
    k_gemm<<<dim3(125, 1, 1), 256, 0, stream>>>(Z0, 224, 0, WcT0, 224, 0,
                                                Cc0, 64, 0, 0, 7, bc0, 1);
    k_update_h<<<4000, 256, 0, stream>>>(H0s, UR0, Cc0);

    k_pack_Z1<<<16000, 256, 0, stream>>>(Gfin, G1fin, H0s, H1s, Z1, t);
    k_gemm<<<dim3(125, 2, 1), 256, 0, stream>>>(Z1, 256, 0, WurT1, 256, 0,
                                                UR1, 128, 0, 0, 8, biasUR1, 2);
    k_patch_Zc1<<<4000, 256, 0, stream>>>(Z1, UR1, H1s);
    k_gemm<<<dim3(125, 1, 1), 256, 0, stream>>>(Z1, 256, 0, WcT1, 256, 0,
                                                Cc1, 64, 0, 0, 8, bc1, 1);
    k_update_h<<<4000, 256, 0, stream>>>(H1s, UR1, Cc1);
  }

  // ---- output head ----
  k_head<<<(R2 + 255) / 256, 256, 0, stream>>>(H1s, wo, bo, wf0, bf0, wf1, bf1,
                                               (float*)d_out);
}
